// SequenceModel_5463198400790
// MI455X (gfx1250) — matrix operand order repaired and checked by reading, not compiled, not run on hardware
//
#include <hip/hip_runtime.h>

// Fused 2-layer LSTM + linear head, persistent kernel for MI455X (gfx1250).
// 16 workgroups x 256 threads (8 wave32). Wave w owns hidden block w (16 units).
// Weights: Whh0, Wih1 -> LDS (bf16); Whh1, Wlin -> register-resident WMMA B tiles.
// Recurrence: v_wmma_f32_16x16x32_bf16; layer-0 input proj: v_wmma_f32_16x16x4_f32.
// x streamed with double-buffered global_load_async_to_lds_b128 (ASYNCcnt).

namespace {

constexpr int Tn  = 2048;
constexpr int Hn  = 128;
constexpr int INn = 4;
constexpr int CH  = 16;          // timesteps per async x chunk
constexpr int NCH = Tn / CH;     // 128 chunks

typedef __attribute__((ext_vector_type(16))) __bf16 v16bf;
typedef __attribute__((ext_vector_type(8)))  __bf16 v8bf;
typedef __attribute__((ext_vector_type(8)))  float  v8f;
typedef __attribute__((ext_vector_type(4)))  float  v4f;
typedef __attribute__((ext_vector_type(2)))  float  v2f;
typedef __attribute__((ext_vector_type(4)))  unsigned short u16x4;

__device__ inline unsigned short f2bfu(float f) {
  unsigned u = __builtin_bit_cast(unsigned, f);
  unsigned r = (u + 0x7FFFu + ((u >> 16) & 1u)) >> 16;  // round-to-nearest-even
  return (unsigned short)r;
}
__device__ inline __bf16 f2bf(float f) {
  return __builtin_bit_cast(__bf16, f2bfu(f));
}

__device__ inline v8f splat8(float x) {
  v8f v;
#pragma unroll
  for (int i = 0; i < 8; ++i) v[i] = x;
  return v;
}
__device__ inline float fsig(float x) {
  return __builtin_amdgcn_rcpf(1.0f + __expf(-x));
}
__device__ inline float ftanh(float x) {
  float e = __expf(2.0f * x);
  return 1.0f - 2.0f * __builtin_amdgcn_rcpf(e + 1.0f);
}
__device__ inline v8f sig8(v8f x) {
  v8f r;
#pragma unroll
  for (int i = 0; i < 8; ++i) r[i] = fsig(x[i]);
  return r;
}
__device__ inline v8f tanh8(v8f x) {
  v8f r;
#pragma unroll
  for (int i = 0; i < 8; ++i) r[i] = ftanh(x[i]);
  return r;
}
__device__ inline v16bf cat16(v8bf lo, v8bf hi) {
  return __builtin_shufflevector(lo, hi, 0,1,2,3,4,5,6,7,8,9,10,11,12,13,14,15);
}

// A tile (16x32 bf16, M x K) from row-major [16][128] LDS h-buffer.
// Lanes 0-15: row=lane, elems = K(0..15); lanes 16-31: K(16..31).
__device__ inline v16bf loadA(const __bf16* h, int lane, int kc) {
  const int row = lane & 15;
  const int kb  = kc * 32 + ((lane & 16) ? 16 : 0);
  v8bf lo = *(const v8bf*)(h + row * Hn + kb);
  v8bf hi = *(const v8bf*)(h + row * Hn + kb + 8);
  return cat16(lo, hi);
}

// B tile (32x16 bf16, K x N); column n of B = row `wrow` of the row-major weight.
// Lanes 0-15 hold K 0..15, lanes 16-31 hold K 16..31 (contiguous in the row).
__device__ inline v16bf loadB(const __bf16* w, int wrow, int kc, int lane) {
  const int kb = kc * 32 + ((lane & 16) ? 16 : 0);
  v8bf lo = *(const v8bf*)(w + wrow * Hn + kb);
  v8bf hi = *(const v8bf*)(w + wrow * Hn + kb + 8);
  return cat16(lo, hi);
}

__device__ inline v8f wmma_bf16(v16bf a, v16bf b, v8f c) {
  return __builtin_amdgcn_wmma_f32_16x16x32_bf16(false, a, false, b, (short)0, c,
                                                 false, false);
}

} // namespace

__global__ __launch_bounds__(256, 1)
void lstm_fused_kernel(const float* __restrict__ x,
                       const float* __restrict__ Wih0, const float* __restrict__ Whh0,
                       const float* __restrict__ bih0, const float* __restrict__ bhh0,
                       const float* __restrict__ Wih1, const float* __restrict__ Whh1,
                       const float* __restrict__ bih1, const float* __restrict__ bhh1,
                       const float* __restrict__ Wlin, const float* __restrict__ blin,
                       float* __restrict__ out_v, float* __restrict__ out_z)
{
  __shared__ __align__(16) __bf16 sWhh0[4 * Hn * Hn];   // 128 KB
  __shared__ __align__(16) __bf16 sWih1[4 * Hn * Hn];   // 128 KB
  __shared__ __align__(16) __bf16 sH0[16 * Hn];         // 4 KB, layer-0 hidden
  __shared__ __align__(16) __bf16 sH1[16 * Hn];         // 4 KB, layer-1 hidden
  __shared__ __align__(16) float  sX[2][16][CH][INn];   // 8 KB, x double buffer

  const int tid  = threadIdx.x;
  const int lane = tid & 31;
  const int hb   = tid >> 5;          // wave id == hidden block (8 x 16 units)
  const int nL   = lane & 15;         // N within 16-wide tile
  const int hi8  = (lane & 16) ? 8 : 0;
  const int b0   = blockIdx.x * 16;   // batch tile base

  // one b128 per thread per chunk: 16 rows x 16 steps x 16B = 4 KB
  auto issueChunk = [&](int c) {
    const int row = tid >> 4;   // batch row in tile
    const int seg = tid & 15;   // step within chunk
    const float* gp = x + ((size_t)(b0 + row) * Tn + c * CH + seg) * INn;
    unsigned lds_off = (unsigned)(uintptr_t)&sX[c & 1][row][seg][0];
    unsigned long long ga = (unsigned long long)(uintptr_t)gp;
    asm volatile("global_load_async_to_lds_b128 %0, %1, off"
                 :: "v"(lds_off), "v"(ga) : "memory");
  };

  issueChunk(0);  // overlap first x chunk with weight staging

  // ---- one-time stage of the K=128 weight matrices into LDS as bf16 ----
#pragma unroll 1
  for (int i = tid * 4; i < 4 * Hn * Hn; i += 256 * 4) {
    v4f w0 = *(const v4f*)(Whh0 + i);
    v4f w1 = *(const v4f*)(Wih1 + i);
    *(u16x4*)(sWhh0 + i) = (u16x4){f2bfu(w0.x), f2bfu(w0.y), f2bfu(w0.z), f2bfu(w0.w)};
    *(u16x4*)(sWih1 + i) = (u16x4){f2bfu(w1.x), f2bfu(w1.y), f2bfu(w1.z), f2bfu(w1.w)};
  }
  for (int i = tid; i < 16 * Hn; i += 256) {  // h(-1) = 0
    sH0[i] = f2bf(0.0f);
    sH1[i] = f2bf(0.0f);
  }

  // ---- register-resident B tiles ----
  const int koffB = (lane & 16) ? 16 : 0;
  v16bf wr[4][4];  // Whh1, per gate g, K-chunk kc
#pragma unroll
  for (int g = 0; g < 4; ++g)
#pragma unroll
    for (int kc = 0; kc < 4; ++kc) {
      const float* src = Whh1 + (size_t)(g * Hn + hb * 16 + nL) * Hn + kc * 32 + koffB;
      v16bf t;
#pragma unroll
      for (int i = 0; i < 16; ++i) t[i] = f2bf(src[i]);
      wr[g][kc] = t;
    }
  v16bf wl[4];     // Wlin head, columns >= 4 zero-padded
#pragma unroll
  for (int kc = 0; kc < 4; ++kc) {
    v16bf t;
#pragma unroll
    for (int i = 0; i < 16; ++i)
      t[i] = (nL < 4) ? f2bf(Wlin[(size_t)nL * Hn + kc * 32 + koffB + i]) : f2bf(0.0f);
    wl[kc] = t;
  }
  // Wih0 (512x4) as f32 B tiles for the 16x16x4 WMMA; fused biases.
  v2f   bw0[4];
  float bias0[4], bias1[4];
#pragma unroll
  for (int g = 0; g < 4; ++g) {
    const int col = g * Hn + hb * 16 + nL;
    const float* s = Wih0 + (size_t)col * INn + ((lane & 16) ? 2 : 0);
    bw0[g]   = (v2f){s[0], s[1]};
    bias0[g] = bih0[col] + bhh0[col];
    bias1[g] = bih1[col] + bhh1[col];
  }
  const float blv = (nL < 4) ? blin[nL] : 0.0f;

  __syncthreads();

  v8f c0 = splat8(0.0f), c1 = splat8(0.0f);

#pragma unroll 1
  for (int t = 0; t < Tn; ++t) {
    // ---- x chunk pipeline (double buffered, ASYNCcnt ordered in-order) ----
    if ((t & (CH - 1)) == 0) {
      const int c = t >> 4;
      if (c + 1 < NCH) {
        issueChunk(c + 1);  // target buffer held chunk c-1: fully read by now
        asm volatile("s_wait_asynccnt 0x1" ::: "memory");  // chunk c has landed
      } else {
        asm volatile("s_wait_asynccnt 0x0" ::: "memory");
      }
      __syncthreads();  // make every thread's async writes visible block-wide
    }

    // ===== layer 0: g = x Wih0^T + h0 Whh0^T + b =====
    const float* xr = &sX[(t >> 4) & 1][nL][t & (CH - 1)][(lane & 16) ? 2 : 0];
    v2f xa = (v2f){xr[0], xr[1]};  // A tile 16x4 f32
    v8f acc[4];
#pragma unroll
    for (int g = 0; g < 4; ++g) {
      acc[g] = splat8(bias0[g]);
      acc[g] = __builtin_amdgcn_wmma_f32_16x16x4_f32(false, xa, false, bw0[g],
                                                     (short)0, acc[g], false, false);
    }
#pragma unroll
    for (int kc = 0; kc < 4; ++kc) {
      v16bf a = loadA(sH0, lane, kc);
#pragma unroll
      for (int g = 0; g < 4; ++g)
        acc[g] = wmma_bf16(a, loadB(sWhh0, g * Hn + hb * 16 + nL, kc, lane), acc[g]);
    }
    __syncthreads();  // all sH0(t-1) readers done before overwrite

    v8f iv = sig8(acc[0]);
    v8f fv = sig8(acc[1]);
    v8f gv = tanh8(acc[2]);
    v8f ov = sig8(acc[3]);
    c0 = fv * c0 + iv * gv;
    v8f h0v = ov * tanh8(c0);
#pragma unroll
    for (int r = 0; r < 8; ++r)
      sH0[(r + hi8) * Hn + hb * 16 + nL] = f2bf(h0v[r]);
    __syncthreads();  // sH0(t) visible

    // ===== layer 1: g = h0 Wih1^T + h1 Whh1^T + b =====
#pragma unroll
    for (int g = 0; g < 4; ++g) acc[g] = splat8(bias1[g]);
#pragma unroll
    for (int kc = 0; kc < 4; ++kc) {
      v16bf a0 = loadA(sH0, lane, kc);
      v16bf a1 = loadA(sH1, lane, kc);
#pragma unroll
      for (int g = 0; g < 4; ++g) {
        acc[g] = wmma_bf16(a0, loadB(sWih1, g * Hn + hb * 16 + nL, kc, lane), acc[g]);
        acc[g] = wmma_bf16(a1, wr[g][kc], acc[g]);
      }
    }
    __syncthreads();  // all sH1(t-1) readers done before overwrite

    iv = sig8(acc[0]);
    fv = sig8(acc[1]);
    gv = tanh8(acc[2]);
    ov = sig8(acc[3]);
    c1 = fv * c1 + iv * gv;
    v8f h1v = ov * tanh8(c1);
#pragma unroll
    for (int r = 0; r < 8; ++r)
      sH1[(r + hi8) * Hn + hb * 16 + nL] = f2bf(h1v[r]);
    if (t == Tn - 1) {  // z = zs[:, -1] (f32, straight from registers)
#pragma unroll
      for (int r = 0; r < 8; ++r)
        out_z[(size_t)(b0 + r + hi8) * Hn + hb * 16 + nL] = h1v[r];
    }
    __syncthreads();  // sH1(t) visible

    // ===== head: v[:, t] = h1 Wlin^T + blin (wave 0 only; EXEC all-ones) =====
    if (hb == 0) {
      v8f av = splat8(blv);
#pragma unroll
      for (int kc = 0; kc < 4; ++kc)
        av = wmma_bf16(loadA(sH1, lane, kc), wl[kc], av);
      if (nL < 4) {
#pragma unroll
        for (int r = 0; r < 8; ++r)
          out_v[(((size_t)(b0 + r + hi8)) * Tn + t) * 4 + nL] = av[r];
      }
    }
  }
}

extern "C" void kernel_launch(void* const* d_in, const int* in_sizes, int n_in,
                              void* d_out, int out_size, void* d_ws, size_t ws_size,
                              hipStream_t stream) {
  (void)in_sizes; (void)n_in; (void)out_size; (void)d_ws; (void)ws_size;
  const float* x    = (const float*)d_in[0];
  const float* Wih0 = (const float*)d_in[1];
  const float* Whh0 = (const float*)d_in[2];
  const float* bih0 = (const float*)d_in[3];
  const float* bhh0 = (const float*)d_in[4];
  const float* Wih1 = (const float*)d_in[5];
  const float* Whh1 = (const float*)d_in[6];
  const float* bih1 = (const float*)d_in[7];
  const float* bhh1 = (const float*)d_in[8];
  const float* Wlin = (const float*)d_in[9];
  const float* blin = (const float*)d_in[10];

  float* out_v = (float*)d_out;                       // [256, 2048, 4]
  float* out_z = out_v + (size_t)256 * 2048 * 4;      // [256, 128]

  lstm_fused_kernel<<<dim3(16), dim3(256), 0, stream>>>(
      x, Wih0, Whh0, bih0, bhh0, Wih1, Whh1, bih1, bhh1, Wlin, blin,
      out_v, out_z);
}
